// CausalScaledDotAttention_6502580486603
// MI455X (gfx1250) — matrix operand order repaired and checked by reading, not compiled, not run on hardware
//
#include <hip/hip_runtime.h>
#include <hip/hip_bf16.h>
#include <stdint.h>

#define B_ 8
#define S_ 2048
#define H_ 1024
#define NEG_INF -10000000.0f

typedef __attribute__((ext_vector_type(16))) __bf16          v16bf;
typedef __attribute__((ext_vector_type(16))) unsigned short  v16us;
typedef __attribute__((ext_vector_type(8)))  float           v8f;
typedef __attribute__((ext_vector_type(8)))  unsigned int    v8u;

#if __has_builtin(__builtin_amdgcn_global_load_async_to_lds_b128)
#define USE_ASYNC_LDS 1
// Exact param types per hipcc diagnostic: vector_size(16) int, AS1 / AS3.
typedef int v4i_vs __attribute__((vector_size(16)));
typedef __attribute__((address_space(1))) v4i_vs* gbl_v4p;
typedef __attribute__((address_space(3))) v4i_vs* lds_v4p;
#endif

static __device__ __forceinline__ void wait_async0() {
#if __has_builtin(__builtin_amdgcn_s_wait_asynccnt)
    __builtin_amdgcn_s_wait_asynccnt(0);
#else
    asm volatile("s_wait_asynccnt 0" ::: "memory");
#endif
}

static __device__ __forceinline__ unsigned short f32_bf16(float f) {
    unsigned int u = __float_as_uint(f);
    u += 0x7FFFu + ((u >> 16) & 1u);          // round-to-nearest-even
    return (unsigned short)(u >> 16);
}

static __device__ __forceinline__ v8f wmma_bf16(v16bf a, v16bf b, v8f c) {
    return __builtin_amdgcn_wmma_f32_16x16x32_bf16(false, a, false, b,
                                                   (short)0, c, false, false);
}

// ---------------------------------------------------------------- cvt
__global__ void cvt_f32_bf16(const float* __restrict__ in,
                             unsigned short* __restrict__ out, int n) {
    for (int i = blockIdx.x * blockDim.x + threadIdx.x; i < n;
         i += gridDim.x * blockDim.x)
        out[i] = f32_bf16(in[i]);
}

// ---------------------------------------------------------------- fragments
// A fragment (16x32, MxK): lane m = lane&15, kh = lane>>4.
// elem j -> K = 16*kh + j. Two b128 loads.
static __device__ __forceinline__ v16bf frag_a(const unsigned short* __restrict__ p,
                                               int ldk, int lane) {
    int m = lane & 15, kh = lane >> 4;
    const uint4* q0 = (const uint4*)(p + (size_t)m * ldk + 16 * kh);
    const uint4* q1 = (const uint4*)(p + (size_t)m * ldk + 16 * kh + 8);
    uint4 a0 = *q0, a1 = *q1;
    v8u u = {a0.x, a0.y, a0.z, a0.w, a1.x, a1.y, a1.z, a1.w};
    return __builtin_bit_cast(v16bf, u);
}
// B fragment (32x16, KxN) where B_op[k][n] = Wrow[n][k], W row-major [N x K].
// lane n = lane&15, kh = lane>>4; elem j -> K = 16*kh + j. Two b128 loads.
static __device__ __forceinline__ v16bf frag_b(const unsigned short* __restrict__ p,
                                               int ldk, int lane) {
    int n = lane & 15, kh = lane >> 4;
    const uint4* q0 = (const uint4*)(p + (size_t)n * ldk + 16 * kh);
    uint4 b0 = q0[0], b1 = q0[1];
    v8u u = {b0.x, b0.y, b0.z, b0.w, b1.x, b1.y, b1.z, b1.w};
    return __builtin_bit_cast(v16bf, u);
}

// ---------------------------------------------------------------- projections
// O[M x N] (bf16) = X[M x K] @ W[N x K]^T + bias.
// Register-blocked: one 32(M) x 64(N) tile per wave: 2 A-frags x 4 B-frags,
// 8 wmma per 12 b128 loads per K-step.
__global__ void proj_gemm_bf16(const unsigned short* __restrict__ X,
                               const unsigned short* __restrict__ W,
                               const float* __restrict__ bias,
                               unsigned short* __restrict__ O,
                               int M, int N, int K) {
    int wave = blockIdx.x * (blockDim.x >> 5) + (threadIdx.x >> 5);
    int lane = threadIdx.x & 31;
    int ntiles = N >> 6;
    int tm = wave / ntiles, tn = wave % ntiles;
    if (tm >= (M >> 5)) return;
    const unsigned short* a0 = X + (size_t)(tm * 32) * K;
    const unsigned short* a1 = a0 + (size_t)16 * K;
    const unsigned short* bb = W + (size_t)(tn * 64) * K;
    v8f c[8] = {};
    for (int k = 0; k < K; k += 32) {
        v16bf A0 = frag_a(a0 + k, K, lane);
        v16bf A1 = frag_a(a1 + k, K, lane);
        v16bf Bf0 = frag_b(bb + k, K, lane);
        v16bf Bf1 = frag_b(bb + (size_t)16 * K + k, K, lane);
        v16bf Bf2 = frag_b(bb + (size_t)32 * K + k, K, lane);
        v16bf Bf3 = frag_b(bb + (size_t)48 * K + k, K, lane);
        c[0] = wmma_bf16(A0, Bf0, c[0]);
        c[1] = wmma_bf16(A0, Bf1, c[1]);
        c[2] = wmma_bf16(A0, Bf2, c[2]);
        c[3] = wmma_bf16(A0, Bf3, c[3]);
        c[4] = wmma_bf16(A1, Bf0, c[4]);
        c[5] = wmma_bf16(A1, Bf1, c[5]);
        c[6] = wmma_bf16(A1, Bf2, c[6]);
        c[7] = wmma_bf16(A1, Bf3, c[7]);
    }
    int ln = lane & 15, mh = lane >> 4;
#pragma unroll
    for (int mi = 0; mi < 2; ++mi)
#pragma unroll
        for (int ni = 0; ni < 4; ++ni) {
            int n = tn * 64 + ni * 16 + ln;
            float bv = bias[n];
#pragma unroll
            for (int r = 0; r < 8; ++r) {
                int m = tm * 32 + mi * 16 + 8 * mh + r;
                O[(size_t)m * N + n] = f32_bf16(c[mi * 4 + ni][r] + bv);
            }
        }
}

// ---------------------------------------------------------------- scores
// attn_raw[b,s,t] = (s<=t) ? (k_s . q_t)/32 : NEG_INF
// 32(s) x 64(t) tile per wave; fully-masked tiles skip the GEMM.
__global__ void scores_kernel(const unsigned short* __restrict__ Kb,
                              const unsigned short* __restrict__ Qb,
                              float* __restrict__ attn) {
    const float scale = 0.03125f;  // 1/sqrt(1024)
    int wave = blockIdx.x * (blockDim.x >> 5) + (threadIdx.x >> 5);
    int lane = threadIdx.x & 31;
    const int ST = S_ / 32, TT = S_ / 64;
    int b  = wave / (ST * TT);
    int rr = wave % (ST * TT);
    int ts = rr / TT, tt = rr % TT;
    int s0 = ts * 32, t0 = tt * 64;
    float* outp = attn + (size_t)b * S_ * S_;
    int ln = lane & 15, mh = lane >> 4;
    if (s0 > t0 + 63) {  // fully masked: just fill
#pragma unroll
        for (int mi = 0; mi < 2; ++mi)
#pragma unroll
            for (int ni = 0; ni < 4; ++ni)
#pragma unroll
                for (int r = 0; r < 8; ++r)
                    outp[(size_t)(s0 + mi * 16 + 8 * mh + r) * S_ +
                         t0 + ni * 16 + ln] = NEG_INF;
        return;
    }
    const unsigned short* a0 = Kb + ((size_t)b * S_ + s0) * H_;
    const unsigned short* a1 = a0 + (size_t)16 * H_;
    const unsigned short* qb = Qb + ((size_t)b * S_ + t0) * H_;
    v8f c[8] = {};
    for (int h = 0; h < H_; h += 32) {
        v16bf A0 = frag_a(a0 + h, H_, lane);
        v16bf A1 = frag_a(a1 + h, H_, lane);
        v16bf Bf0 = frag_b(qb + h, H_, lane);
        v16bf Bf1 = frag_b(qb + (size_t)16 * H_ + h, H_, lane);
        v16bf Bf2 = frag_b(qb + (size_t)32 * H_ + h, H_, lane);
        v16bf Bf3 = frag_b(qb + (size_t)48 * H_ + h, H_, lane);
        c[0] = wmma_bf16(A0, Bf0, c[0]);
        c[1] = wmma_bf16(A0, Bf1, c[1]);
        c[2] = wmma_bf16(A0, Bf2, c[2]);
        c[3] = wmma_bf16(A0, Bf3, c[3]);
        c[4] = wmma_bf16(A1, Bf0, c[4]);
        c[5] = wmma_bf16(A1, Bf1, c[5]);
        c[6] = wmma_bf16(A1, Bf2, c[6]);
        c[7] = wmma_bf16(A1, Bf3, c[7]);
    }
#pragma unroll
    for (int mi = 0; mi < 2; ++mi)
#pragma unroll
        for (int ni = 0; ni < 4; ++ni) {
            int t = t0 + ni * 16 + ln;
#pragma unroll
            for (int r = 0; r < 8; ++r) {
                int s = s0 + mi * 16 + 8 * mh + r;
                float v = c[mi * 4 + ni][r] * scale;
                if (s > t) v = NEG_INF;
                outp[(size_t)s * S_ + t] = v;
            }
        }
}

// ---------------------------------------------------------------- softmax over s
// 64 columns per block, 4 threads per column (s split into quarters),
// LDS reduction for max and sum. Row reads are 256B contiguous.
__global__ void softmax_col(float* __restrict__ attn) {
    __shared__ float red[256];
    int tid = threadIdx.x;
    int tcol = tid & 63, q = tid >> 6;
    int b  = blockIdx.x / (S_ / 64);
    int tc = blockIdx.x % (S_ / 64);
    int t = tc * 64 + tcol;
    float* p = attn + (size_t)b * S_ * S_ + t;
    int sBeg = q * (S_ / 4), sEnd = sBeg + (S_ / 4);
    float m = -3.0e38f;
    for (int s = sBeg; s < sEnd; ++s) m = fmaxf(m, p[(size_t)s * S_]);
    red[tid] = m;
    __syncthreads();
    m = fmaxf(fmaxf(red[tcol], red[tcol + 64]),
              fmaxf(red[tcol + 128], red[tcol + 192]));
    __syncthreads();
    float sum = 0.f;
    for (int s = sBeg; s < sEnd; ++s) {
        float e = __expf(p[(size_t)s * S_] - m);
        sum += e;
        p[(size_t)s * S_] = e;
    }
    red[tid] = sum;
    __syncthreads();
    sum = (red[tcol] + red[tcol + 64]) + (red[tcol + 128] + red[tcol + 192]);
    float inv = 1.0f / sum;
    for (int s = sBeg; s < sEnd; ++s) p[(size_t)s * S_] *= inv;
}

// ---------------------------------------------------------------- context
// ctx[b,t,h] = sum_s P[b,s,t] * v[b,s,h].
// Block: 128 thr = 4 waves computes 16(t) x 128(h); each wave 16x32 (2 acc).
// Double-buffered LDS staging: tile i+1's async global->LDS DMA (ASYNCcnt)
// and P fp32->bf16 conversion overlap tile i's fragment builds + WMMAs;
// s_wait_asynccnt 0 drains the prefetch only after compute.
__global__ void context_kernel(const float* __restrict__ attn,
                               const unsigned short* __restrict__ Vb,
                               float* __restrict__ ctx) {
    __shared__ unsigned short Pl[2][32 * 16];
    __shared__ unsigned short Vl[2][32 * 128];
    int wid = threadIdx.x >> 5, lane = threadIdx.x & 31;
    const int TT = S_ / 16, HS = H_ / 128;
    int b  = blockIdx.x / (TT * HS);
    int rr = blockIdx.x % (TT * HS);
    int tt = rr / HS, hs = rr % HS;
    int t0 = tt * 16, h0 = hs * 128;
    const float* Pp = attn + (size_t)b * S_ * S_;
    const unsigned short* Vp = Vb + (size_t)b * S_ * H_ + h0;
    int n = lane & 15, kh = lane >> 4;
    v8f c0 = {}, c1 = {};

    // ---- prologue: stage tile 0 into buffer 0
#ifdef USE_ASYNC_LDS
    for (int i = threadIdx.x; i < 512; i += blockDim.x) {
        int s = i >> 4, c8 = i & 15;
        __builtin_amdgcn_global_load_async_to_lds_b128(
            (gbl_v4p)(Vp + (size_t)s * H_ + c8 * 8),
            (lds_v4p)(&Vl[0][s * 128 + c8 * 8]), 0, 0);
    }
#else
    for (int i = threadIdx.x; i < 4096; i += blockDim.x)
        Vl[0][i] = Vp[(size_t)(i >> 7) * H_ + (i & 127)];
#endif
    for (int i = threadIdx.x; i < 512; i += blockDim.x)
        Pl[0][i] = f32_bf16(Pp[(size_t)(i >> 4) * S_ + t0 + (i & 15)]);
#ifdef USE_ASYNC_LDS
    wait_async0();
#endif
    __syncthreads();

    for (int k0 = 0; k0 < S_; k0 += 32) {
        int buf = (k0 >> 5) & 1;
        int nxt = buf ^ 1;
        bool more = (k0 + 32) < S_;
        // ---- prefetch tile i+1 into the alternate buffer (overlaps compute)
        if (more) {
#ifdef USE_ASYNC_LDS
            for (int i = threadIdx.x; i < 512; i += blockDim.x) {
                int s = i >> 4, c8 = i & 15;
                __builtin_amdgcn_global_load_async_to_lds_b128(
                    (gbl_v4p)(Vp + (size_t)(k0 + 32 + s) * H_ + c8 * 8),
                    (lds_v4p)(&Vl[nxt][s * 128 + c8 * 8]), 0, 0);
            }
#else
            for (int i = threadIdx.x; i < 4096; i += blockDim.x)
                Vl[nxt][i] = Vp[(size_t)(k0 + 32 + (i >> 7)) * H_ + (i & 127)];
#endif
            for (int i = threadIdx.x; i < 512; i += blockDim.x)
                Pl[nxt][i] =
                    f32_bf16(Pp[(size_t)(k0 + 32 + (i >> 4)) * S_ + t0 + (i & 15)]);
        }
        // ---- compute on current buffer
        v16us av, bv0, bv1;
#pragma unroll
        for (int j = 0; j < 16; ++j) {
            int sa = 16 * kh + j;
            av[j] = Pl[buf][sa * 16 + n];                  // A_op[t][s] = P[s][t]
            int sb = 16 * kh + j;
            bv0[j] = Vl[buf][sb * 128 + wid * 32 + n];     // B_op[s][h]
            bv1[j] = Vl[buf][sb * 128 + wid * 32 + 16 + n];
        }
        v16bf af = __builtin_bit_cast(v16bf, av);
        c0 = wmma_bf16(af, __builtin_bit_cast(v16bf, bv0), c0);
        c1 = wmma_bf16(af, __builtin_bit_cast(v16bf, bv1), c1);
#ifdef USE_ASYNC_LDS
        if (more) wait_async0();
#endif
        __syncthreads();
    }
    int mh = lane >> 4;
#pragma unroll
    for (int r = 0; r < 8; ++r) {
        int t = t0 + 8 * mh + r;
        float* row = ctx + ((size_t)b * S_ + t) * H_ + h0 + wid * 32;
        row[n] = c0[r];
        row[16 + n] = c1[r];
    }
}

// ---------------------------------------------------------------- launch
extern "C" void kernel_launch(void* const* d_in, const int* in_sizes, int n_in,
                              void* d_out, int out_size, void* d_ws, size_t ws_size,
                              hipStream_t stream) {
    const float* queries = (const float*)d_in[0];
    const float* keys    = (const float*)d_in[1];
    const float* values  = (const float*)d_in[2];
    const float* Wq      = (const float*)d_in[3];
    const float* bq      = (const float*)d_in[4];
    const float* Wk      = (const float*)d_in[5];
    const float* bk      = (const float*)d_in[6];
    const float* Wv      = (const float*)d_in[7];
    const float* bv      = (const float*)d_in[8];

    const size_t MH = (size_t)B_ * S_ * H_;   // 16,777,216
    const size_t HH = (size_t)H_ * H_;

    float* ctx  = (float*)d_out;              // [B,S,H]
    float* attn = ctx + MH;                   // [B,S,S]

    unsigned short* xq = (unsigned short*)d_ws;
    unsigned short* xk = xq + MH;
    unsigned short* xv = xk + MH;
    unsigned short* wq = xv + MH;
    unsigned short* wk = wq + HH;
    unsigned short* wv = wk + HH;
    unsigned short* qb = wv + HH;
    unsigned short* kb = qb + MH;
    unsigned short* vb = kb + MH;

    // 1) fp32 -> bf16
    cvt_f32_bf16<<<2048, 256, 0, stream>>>(queries, xq, (int)MH);
    cvt_f32_bf16<<<2048, 256, 0, stream>>>(keys,    xk, (int)MH);
    cvt_f32_bf16<<<2048, 256, 0, stream>>>(values,  xv, (int)MH);
    cvt_f32_bf16<<<1024, 256, 0, stream>>>(Wq, wq, (int)HH);
    cvt_f32_bf16<<<1024, 256, 0, stream>>>(Wk, wk, (int)HH);
    cvt_f32_bf16<<<1024, 256, 0, stream>>>(Wv, wv, (int)HH);

    // 2) projections: (M/32)*(N/64) = 8192 wave-tiles, 4 waves/block
    const int M = B_ * S_;
    int projBlocks = (M / 32) * (H_ / 64) / 4;   // 2048
    proj_gemm_bf16<<<projBlocks, 128, 0, stream>>>(xq, wq, bq, qb, M, H_, H_);
    proj_gemm_bf16<<<projBlocks, 128, 0, stream>>>(xk, wk, bk, kb, M, H_, H_);
    proj_gemm_bf16<<<projBlocks, 128, 0, stream>>>(xv, wv, bv, vb, M, H_, H_);

    // 3) masked raw scores: B*(S/32)*(S/64) = 16384 wave-tiles / 4 per block
    scores_kernel<<<16384 / 4, 128, 0, stream>>>(kb, qb, attn);

    // 4) column softmax (in place on d_out attn region)
    softmax_col<<<B_ * (S_ / 64), 256, 0, stream>>>(attn);

    // 5) context = P^T @ V : B*(S/16)*(H/128) = 8192 blocks
    context_kernel<<<B_ * (S_ / 16) * (H_ / 128), 128, 0, stream>>>(attn, vb, ctx);
}
